// CustomGRU_79259326480850
// MI455X (gfx1250) — compile-verified, matrix operand order checked by reading, not run on hardware
//
#include <hip/hip_runtime.h>
#include <hip/hip_bf16.h>

#define Bb 256
#define Tt 256
#define Dd 512
#define Hh 512
#define Gg 1536  // 3*H

typedef __attribute__((ext_vector_type(16))) __bf16 v16bf;
typedef __attribute__((ext_vector_type(8)))  float  v8f;
typedef __attribute__((ext_vector_type(4)))  unsigned int u32x4;
typedef __attribute__((ext_vector_type(8)))  int i32x8;
typedef __attribute__((ext_vector_type(4)))  int i32x4;

// ---------------------------------------------------------------------------
// WMMA helpers (CDNA5 16x16x32 bf16, wave32)
// ---------------------------------------------------------------------------
__device__ __forceinline__ v8f wmma_bf16(v16bf a, v16bf b, v8f c) {
  return __builtin_amdgcn_wmma_f32_16x16x32_bf16(false, a, false, b, (short)0, c,
                                                 false, false);
}

// 16x32 bf16 fragment load (A layout; same layout works as B for X@W^T since
// both operands are K-contiguous row-major). Works for global or LDS source
// (addrspace inferred after inlining -> global_load_b128 / ds_load_b128).
__device__ __forceinline__ v16bf ld_frag(const __bf16* src, int ld, int row0, int k0) {
  int l = threadIdx.x & 31;
  const __bf16* p = src + (size_t)(row0 + (l & 15)) * ld + (k0 + ((l >> 4) << 3));
  union { v16bf v; uint4 q[2]; } u;
  u.q[0] = *(const uint4*)(p);
  u.q[1] = *(const uint4*)(p + 16);
  return u.v;
}

// ---------------------------------------------------------------------------
// Tensor Data Mover: 2D tile (rows x cols bf16, row-major, stride in elems)
// global -> LDS.  D# layout per CDNA5 ISA.
// Must be called with wave-uniform arguments.
// ---------------------------------------------------------------------------
__device__ __forceinline__ void tdm_load_2d(const void* gbase, unsigned lds_off,
                                            unsigned rows, unsigned cols,
                                            unsigned stride_elems) {
  unsigned long long ga = (unsigned long long)(uintptr_t)gbase;
  u32x4 g0;
  g0[0] = 1u;                                   // count=1 (valid), user mode
  g0[1] = lds_off;                              // lds_addr (byte offset)
  g0[2] = (unsigned)(ga & 0xFFFFFFFFu);         // global_addr[31:0]
  g0[3] = (unsigned)((ga >> 32) & 0x1FFFFFFu)   // global_addr[56:32]
          | (2u << 30);                         // type=2 ("image")
  i32x8 g1;
  g1[0] = (int)(1u << 16);                      // data_size=1 (2 bytes)
  g1[1] = (int)((cols & 0xFFFFu) << 16);        // tensor_dim0[15:0] @ bits 63:48
  g1[2] = (int)((cols >> 16) | ((rows & 0xFFFFu) << 16));   // dim0 hi | tensor_dim1 lo
  g1[3] = (int)((rows >> 16) | ((cols & 0xFFFFu) << 16));   // dim1 hi | tile_dim0
  g1[4] = (int)(rows & 0xFFFFu);                // tile_dim1 (tile_dim2 = 0)
  g1[5] = (int)stride_elems;                    // tensor_dim0_stride[31:0]
  g1[6] = 0;                                    // stride hi / dim1_stride lo
  g1[7] = 0;                                    // dim1_stride hi
  i32x4 z4 = {0, 0, 0, 0};                      // groups 2/3 unused (2D tensor)
  i32x8 z8 = {0, 0, 0, 0, 0, 0, 0, 0};
  __builtin_amdgcn_tensor_load_to_lds(g0, g1, z4, z4, z8, 0);
}

// ---------------------------------------------------------------------------
// Prep kernels
// ---------------------------------------------------------------------------
__global__ void cvt_bf16(const float* __restrict__ s, __bf16* __restrict__ d, int n) {
  int i = blockIdx.x * blockDim.x + threadIdx.x;
  if (i < n) d[i] = (__bf16)s[i];
}

// wcat[n, 0:512] = w_hh[2H+n, :]; wcat[n, 512:1024] = w_tf[n, :]
__global__ void build_wcat(const float* __restrict__ whh, const float* __restrict__ wtf,
                           __bf16* __restrict__ wcat) {
  int i = blockIdx.x * blockDim.x + threadIdx.x;  // Hh * 1024
  int n = i >> 10, k = i & 1023;
  float v = (k < Hh) ? whh[(size_t)(2 * Hh + n) * Hh + k]
                     : wtf[(size_t)n * Hh + (k - Hh)];
  wcat[i] = (__bf16)v;
}

__global__ void init_h(float* __restrict__ hx, __bf16* __restrict__ hbf) {
  int i = blockIdx.x * blockDim.x + threadIdx.x;  // Bb*Hh
  int b = i >> 9, g = i & 511;
  hx[(size_t)b * (Tt + 1) * Hh + g] = 0.0f;
  hbf[i] = (__bf16)0.0f;
}

// ---------------------------------------------------------------------------
// Kernel 1: xg = inputs @ W_ih^T + bias   (M=65536, K=512, N=1536), f32 out
// ---------------------------------------------------------------------------
__global__ __launch_bounds__(128) void gemm_xg(const __bf16* __restrict__ A,
                                               const __bf16* __restrict__ W,
                                               const float* __restrict__ bias,
                                               float* __restrict__ xg) {
  int wg = blockIdx.x * 4 + (threadIdx.x >> 5);
  int mt = (wg & 2047) << 5;
  int nt = (wg >> 11) << 5;
  int l = threadIdx.x & 31;
  int n0 = nt + (l & 15);
  float b0 = bias[n0], b1 = bias[n0 + 16];
  v8f c00, c01, c10, c11;
#pragma unroll
  for (int v = 0; v < 8; ++v) { c00[v] = b0; c01[v] = b1; c10[v] = b0; c11[v] = b1; }
  for (int k = 0; k < Dd; k += 32) {
    if (k + 32 < Dd) {
      __builtin_prefetch(A + (size_t)(mt + (l & 15)) * Dd + k + 32, 0, 0);
      __builtin_prefetch(W + (size_t)(nt + (l & 15)) * Dd + k + 32, 0, 0);
    }
    v16bf a0 = ld_frag(A, Dd, mt, k);
    v16bf a1 = ld_frag(A, Dd, mt + 16, k);
    v16bf w0 = ld_frag(W, Dd, nt, k);
    v16bf w1 = ld_frag(W, Dd, nt + 16, k);
    c00 = wmma_bf16(a0, w0, c00);
    c01 = wmma_bf16(a0, w1, c01);
    c10 = wmma_bf16(a1, w0, c10);
    c11 = wmma_bf16(a1, w1, c11);
  }
  int mb = mt + ((l >> 4) << 3);
#pragma unroll
  for (int v = 0; v < 8; ++v) {
    size_t r0 = (size_t)(mb + v) * Gg;
    size_t r1 = (size_t)(mb + 16 + v) * Gg;
    xg[r0 + n0] = c00[v];
    xg[r0 + n0 + 16] = c01[v];
    xg[r1 + n0] = c10[v];
    xg[r1 + n0 + 16] = c11[v];
  }
}

// ---------------------------------------------------------------------------
// Step kernel A: rz = sigmoid(xg_rz[t] + h @ w_rz^T)
//   M=B=256, N=2H=1024, K=H=512.  Block = 32(M) x 128(N); TDM stages the
//   shared h tile (32x512) and the w_rz strip (128x512) into LDS.
// ---------------------------------------------------------------------------
__global__ __launch_bounds__(128) void step_rz(const float* __restrict__ xg,
                                               const __bf16* __restrict__ hbf,
                                               const __bf16* __restrict__ wrz,
                                               const float* __restrict__ hx,
                                               const float* __restrict__ targets,
                                               __bf16* __restrict__ acat,
                                               float* __restrict__ zbuf, int t) {
  extern __shared__ __bf16 smem[];
  __bf16* sh_h = smem;            // 32 x 512 bf16  (32 KB)
  __bf16* sh_w = smem + 32 * Hh;  // 128 x 512 bf16 (128 KB)
  int warp = threadIdx.x >> 5;
  int mtb = (blockIdx.x & 7) << 5;   // 8 M-blocks of 32
  int ntb = (blockIdx.x >> 3) << 7;  // 8 N-blocks of 128
  if (warp == 0) {
    tdm_load_2d(hbf + (size_t)mtb * Hh, (unsigned)(uintptr_t)sh_h, 32u, (unsigned)Hh,
                (unsigned)Hh);
    tdm_load_2d(wrz + (size_t)ntb * Hh, (unsigned)(uintptr_t)sh_w, 128u, (unsigned)Hh,
                (unsigned)Hh);
    __builtin_amdgcn_s_wait_tensorcnt(0);
  }
  __syncthreads();

  int nt = ntb + (warp << 5);
  int l = threadIdx.x & 31;
  int n0 = nt + (l & 15);
  int mb = mtb + ((l >> 4) << 3);
  const float* cbase = xg + (size_t)t * Gg;
  const size_t cld = (size_t)Tt * Gg;
  v8f c[2][2];
#pragma unroll
  for (int i = 0; i < 2; ++i)
#pragma unroll
    for (int j = 0; j < 2; ++j)
#pragma unroll
      for (int v = 0; v < 8; ++v)
        c[i][j][v] = cbase[(size_t)(mb + 16 * i + v) * cld + n0 + 16 * j];

  int nl = nt - ntb;  // local row in LDS weight strip
  for (int k = 0; k < Hh; k += 32) {
    v16bf a0 = ld_frag(sh_h, Hh, 0, k);
    v16bf a1 = ld_frag(sh_h, Hh, 16, k);
    v16bf w0 = ld_frag(sh_w, Hh, nl, k);
    v16bf w1 = ld_frag(sh_w, Hh, nl + 16, k);
    c[0][0] = wmma_bf16(a0, w0, c[0][0]);
    c[0][1] = wmma_bf16(a0, w1, c[0][1]);
    c[1][0] = wmma_bf16(a1, w0, c[1][0]);
    c[1][1] = wmma_bf16(a1, w1, c[1][1]);
  }

#pragma unroll
  for (int i = 0; i < 2; ++i) {
#pragma unroll
    for (int j = 0; j < 2; ++j) {
      int g = n0 + 16 * j;
#pragma unroll
      for (int v = 0; v < 8; ++v) {
        int b = mb + 16 * i + v;
        float sg = 1.0f / (1.0f + __expf(-c[i][j][v]));
        if (g < Hh) {  // r gate
          float hv = hx[(size_t)b * (Tt + 1) * Hh + (size_t)t * Hh + g];
          float pv = (t == 0) ? 0.0f : targets[((size_t)b * Tt + (t - 1)) * Hh + g];
          acat[(size_t)b * 1024 + g] = (__bf16)(sg * hv);
          acat[(size_t)b * 1024 + Hh + g] = (__bf16)(sg * pv);
        } else {       // z gate
          zbuf[(size_t)b * Hh + (g - Hh)] = sg;
        }
      }
    }
  }
}

// ---------------------------------------------------------------------------
// Step kernel B: s = xg_n[t] + [r*h | r*pt] @ [w_n | w_tf]^T
//   M=256, N=512, K=1024.  Block = 32(M) x 128(N); TDM stages shared acat
//   tile (32x1024, 64 KB) into LDS; weights stream from L2.
// ---------------------------------------------------------------------------
__global__ __launch_bounds__(128) void step_n(const float* __restrict__ xg,
                                              const __bf16* __restrict__ acat,
                                              const __bf16* __restrict__ wcat,
                                              float* __restrict__ sbuf, int t) {
  extern __shared__ __bf16 smem[];
  __bf16* sh_a = smem;  // 32 x 1024 bf16 (64 KB)
  int warp = threadIdx.x >> 5;
  int mtb = (blockIdx.x & 7) << 5;   // 8 M-blocks of 32
  int ntb = (blockIdx.x >> 3) << 7;  // 4 N-blocks of 128
  if (warp == 0) {
    tdm_load_2d(acat + (size_t)mtb * 1024, (unsigned)(uintptr_t)sh_a, 32u, 1024u, 1024u);
    __builtin_amdgcn_s_wait_tensorcnt(0);
  }
  __syncthreads();

  int nt = ntb + (warp << 5);
  int l = threadIdx.x & 31;
  int n0 = nt + (l & 15);
  int mb = mtb + ((l >> 4) << 3);
  const float* cbase = xg + (size_t)t * Gg + 1024;
  const size_t cld = (size_t)Tt * Gg;
  v8f c[2][2];
#pragma unroll
  for (int i = 0; i < 2; ++i)
#pragma unroll
    for (int j = 0; j < 2; ++j)
#pragma unroll
      for (int v = 0; v < 8; ++v)
        c[i][j][v] = cbase[(size_t)(mb + 16 * i + v) * cld + n0 + 16 * j];

  for (int k = 0; k < 1024; k += 32) {
    v16bf a0 = ld_frag(sh_a, 1024, 0, k);
    v16bf a1 = ld_frag(sh_a, 1024, 16, k);
    v16bf w0 = ld_frag(wcat, 1024, nt, k);
    v16bf w1 = ld_frag(wcat, 1024, nt + 16, k);
    c[0][0] = wmma_bf16(a0, w0, c[0][0]);
    c[0][1] = wmma_bf16(a0, w1, c[0][1]);
    c[1][0] = wmma_bf16(a1, w0, c[1][0]);
    c[1][1] = wmma_bf16(a1, w1, c[1][1]);
  }
#pragma unroll
  for (int i = 0; i < 2; ++i)
#pragma unroll
    for (int j = 0; j < 2; ++j)
#pragma unroll
      for (int v = 0; v < 8; ++v)
        sbuf[(size_t)(mb + 16 * i + v) * Hh + n0 + 16 * j] = c[i][j][v];
}

// ---------------------------------------------------------------------------
// Step kernel C: n = softmax(s); hy = (1-z)*n + z*h
// ---------------------------------------------------------------------------
__global__ __launch_bounds__(256) void step_out(const float* __restrict__ sbuf,
                                                const float* __restrict__ zbuf,
                                                float* __restrict__ outputs,
                                                float* __restrict__ hx,
                                                __bf16* __restrict__ hbf, int t) {
  __shared__ float red[256];
  int b = blockIdx.x, tid = threadIdx.x;
  float v0 = sbuf[(size_t)b * Hh + tid];
  float v1 = sbuf[(size_t)b * Hh + tid + 256];
  red[tid] = fmaxf(v0, v1);
  __syncthreads();
  for (int s = 128; s > 0; s >>= 1) {
    if (tid < s) red[tid] = fmaxf(red[tid], red[tid + s]);
    __syncthreads();
  }
  float mx = red[0];
  __syncthreads();
  float e0 = __expf(v0 - mx), e1 = __expf(v1 - mx);
  red[tid] = e0 + e1;
  __syncthreads();
  for (int s = 128; s > 0; s >>= 1) {
    if (tid < s) red[tid] += red[tid + s];
    __syncthreads();
  }
  float inv = 1.0f / red[0];
#pragma unroll
  for (int j = 0; j < 2; ++j) {
    int g = tid + 256 * j;
    float nn = (j ? e1 : e0) * inv;
    float h = hx[(size_t)b * (Tt + 1) * Hh + (size_t)t * Hh + g];
    float z = zbuf[(size_t)b * Hh + g];
    float hy = (1.0f - z) * nn + z * h;
    outputs[((size_t)b * Tt + t) * Hh + g] = hy;
    hx[(size_t)b * (Tt + 1) * Hh + (size_t)(t + 1) * Hh + g] = hy;
    hbf[(size_t)b * Hh + g] = (__bf16)hy;
  }
}

// ---------------------------------------------------------------------------
extern "C" void kernel_launch(void* const* d_in, const int* in_sizes, int n_in,
                              void* d_out, int out_size, void* d_ws, size_t ws_size,
                              hipStream_t stream) {
  (void)in_sizes; (void)n_in; (void)out_size; (void)ws_size;
  const float* inputs  = (const float*)d_in[0];
  const float* targets = (const float*)d_in[1];
  const float* w_ih    = (const float*)d_in[2];
  const float* w_hh    = (const float*)d_in[3];
  const float* b_ih    = (const float*)d_in[4];
  const float* w_tf    = (const float*)d_in[5];

  float* outputs = (float*)d_out;                    // B x T x H
  float* hx = (float*)d_out + (size_t)Bb * Tt * Hh;  // B x (T+1) x H

  char* cur = (char*)d_ws;
  auto alloc = [&](size_t bytes) {
    char* p = cur;
    cur += (bytes + 255) & ~(size_t)255;
    return p;
  };
  float*  xg     = (float*)  alloc((size_t)Bb * Tt * Gg * 4);
  __bf16* in_bf  = (__bf16*) alloc((size_t)Bb * Tt * Dd * 2);
  __bf16* wih_bf = (__bf16*) alloc((size_t)Gg * Dd * 2);
  __bf16* wrz_bf = (__bf16*) alloc((size_t)2 * Hh * Hh * 2);
  __bf16* wcat   = (__bf16*) alloc((size_t)Hh * 2 * Hh * 2);
  __bf16* hbf    = (__bf16*) alloc((size_t)Bb * Hh * 2);
  __bf16* acat   = (__bf16*) alloc((size_t)Bb * 2 * Hh * 2);
  float*  zbuf   = (float*)  alloc((size_t)Bb * Hh * 4);
  float*  sbuf   = (float*)  alloc((size_t)Bb * Hh * 4);

  { int n = Bb * Tt * Dd; cvt_bf16<<<(n + 255) / 256, 256, 0, stream>>>(inputs, in_bf, n); }
  { int n = Gg * Dd;      cvt_bf16<<<(n + 255) / 256, 256, 0, stream>>>(w_ih, wih_bf, n); }
  { int n = 2 * Hh * Hh;  cvt_bf16<<<(n + 255) / 256, 256, 0, stream>>>(w_hh, wrz_bf, n); }
  { int n = Hh * 2 * Hh;  build_wcat<<<(n + 255) / 256, 256, 0, stream>>>(w_hh, w_tf, wcat); }
  { int n = Bb * Hh;      init_h<<<(n + 255) / 256, 256, 0, stream>>>(hx, hbf); }

  gemm_xg<<<24576, 128, 0, stream>>>(in_bf, wih_bf, b_ih, xg);

  const size_t smem_rz = (size_t)(32 * Hh + 128 * Hh) * sizeof(__bf16);  // 160 KB
  const size_t smem_n  = (size_t)(32 * 1024) * sizeof(__bf16);           // 64 KB
  for (int t = 0; t < Tt; ++t) {
    step_rz<<<64, 128, smem_rz, stream>>>(xg, hbf, wrz_bf, hx, targets, acat, zbuf, t);
    step_n<<<32, 128, smem_n, stream>>>(xg, acat, wcat, sbuf, t);
    step_out<<<Bb, 256, 0, stream>>>(sbuf, zbuf, outputs, hx, hbf, t);
  }
}
